// PrxteinMPNN_68616397521365
// MI455X (gfx1250) — matrix operand order repaired and checked by reading, not compiled, not run on hardware
//
#include <hip/hip_runtime.h>
#include <hip/hip_bf16.h>

#define NN 4096
#define KK 32
#define CC 128
#define FFH 512
#define LEn 3
#define LDn 3
#define AOUT 21

typedef __attribute__((ext_vector_type(16))) __bf16 v16bf;
typedef __attribute__((ext_vector_type(16))) unsigned short v16us;
typedef __attribute__((ext_vector_type(8)))  unsigned short v8us;
typedef __attribute__((ext_vector_type(8)))  float v8f;

union Frag16 { v16us u; v16bf b; v8us h2[2]; };

__device__ __forceinline__ unsigned short f2bf(float f) {
  unsigned int u = __float_as_uint(f);
  u += 0x7FFFu + ((u >> 16) & 1u);   // round-to-nearest-even truncation to bf16
  return (unsigned short)(u >> 16);
}

// Native CDNA5 transcendental tanh (one TRANS op instead of libm's ~40-inst
// branchy expansion). v_nop inside the asm covers the TRANS result-use hazard.
__device__ __forceinline__ float fast_tanh(float x) {
  float r;
  asm volatile("v_tanh_f32 %0, %1\n\tv_nop" : "=v"(r) : "v"(x));
  return r;
}

__device__ __forceinline__ float gelu_t(float x) {
  float x3 = x * x * x;
  return 0.5f * x * (1.0f + fast_tanh(0.7978845608028654f * (x + 0.044715f * x3)));
}

// Pack a float4 into 4 bf16 and store as one 8-byte word (dst 8B aligned).
__device__ __forceinline__ void store_bf16x4(unsigned short* dst, float4 v) {
  union { unsigned short s[4]; unsigned long long q; } pk;
  pk.s[0] = f2bf(v.x); pk.s[1] = f2bf(v.y);
  pk.s[2] = f2bf(v.z); pk.s[3] = f2bf(v.w);
  *(unsigned long long*)dst = pk.q;
}

// A fragment: 16x32 bf16 (MxK), row-major source (lda elements, 16B-aligned rows).
// Contiguous k order per lane (matches packed B):
//   lane half h: elements [k0+16h .. k0+16h+15]
__device__ __forceinline__ v16bf load_a_frag(const unsigned short* A, int lda,
                                             int row0, int k0, int lane) {
  Frag16 f;
  int m = lane & 15, h = lane >> 4;
  const v8us* p = (const v8us*)(A + (row0 + m) * lda + k0 + 16 * h);
  f.h2[0] = p[0];
  f.h2[1] = p[1];
  return f.b;
}

// B fragment from pre-swizzled weights: per (kt,nt) 32x16 tile, each lane's 16
// bf16 elements are stored contiguously (32B) -> two coalesced b128 loads.
__device__ __forceinline__ v16bf load_b_packed(const unsigned short* Wp, int nTiles,
                                               int kt, int nt, int lane) {
  Frag16 f;
  const v8us* p = (const v8us*)(Wp + ((((size_t)kt * nTiles + nt) * 32 + lane) << 4));
  f.h2[0] = p[0];
  f.h2[1] = p[1];
  return f.b;
}

// Swizzle fp32 weight W[layers, Kd, NC] into bf16 WMMA-B fragment layout:
// out[(((kt*nTiles + nt)*32 + lane)*16 + e)] = bf16(W[kt*32 + 16*(lane>>4) + e, nt*16 + (lane&15)])
__global__ __launch_bounds__(256) void pack_b_kernel(const float* __restrict__ W,
                                                     unsigned short* __restrict__ out,
                                                     int Kd, int NC, int layers) {
  int per = Kd * NC;
  int total = layers * per;
  int o = blockIdx.x * 256 + threadIdx.x;
  if (o >= total) return;
  int l = o / per;
  int r = o % per;
  int e = r & 15;
  int lane = (r >> 4) & 31;
  int nTiles = NC >> 4;
  int t = r >> 9;
  int nt = t % nTiles;
  int kt = t / nTiles;
  int k = kt * 32 + ((lane >> 4) << 4) + e;
  int n = nt * 16 + (lane & 15);
  out[o] = f2bf(W[(size_t)l * per + (size_t)k * NC + n]);
}

// Fused per-edge 3-layer MLP message kernel. One workgroup per node i; 8 waves,
// each owning a 16-wide N-tile of the 128 output columns.
// IN_C = 384 (encoder: [hV_i, hE_ij, hV_nbr]) or 512 (decoder: [hV_i, hE_ij, 0, hVenc_nbr]).
// reduce=1: out[N,C] = masked mean over K.  reduce=0: out[N,K,C] = LN(hE + m) in-place.
template <int IN_C>
__global__ __launch_bounds__(256) void msg_kernel(
    const float* __restrict__ hVs, const float* __restrict__ hVn,
    const float* __restrict__ hE, const int* __restrict__ nbr,
    const float* __restrict__ mask,
    const unsigned short* __restrict__ W1, const float* __restrict__ b1,
    const unsigned short* __restrict__ W2, const float* __restrict__ b2,
    const unsigned short* __restrict__ W3, const float* __restrict__ b3,
    const float* __restrict__ ln_s, const float* __restrict__ ln_b,
    float* __restrict__ out, int reduce) {
  __shared__ __align__(16) unsigned short xs[KK * IN_C];   // input tile, bf16
  __shared__ __align__(16) unsigned short m1s[KK * CC];    // gelu(x@W1+b1)
  __shared__ __align__(16) unsigned short m2s[KK * CC];    // gelu(m1@W2+b2)
  __shared__ __align__(16) float msf[KK * CC];             // m (f32) for edge-LN path
  __shared__ float mrow[KK];
  __shared__ int   nb[KK];

  const int i = blockIdx.x;
  const int tid = threadIdx.x;
  const int lane = tid & 31;
  const int wv = tid >> 5;
  const int n0 = wv * 16;
  const int n = lane & 15;
  const int h = lane >> 4;

  if (tid < KK) {
    int j = nbr[i * KK + tid];
    nb[tid] = j;
    mrow[tid] = mask[j] * mask[i];
  }
  __syncthreads();

  // Build concat tile (gather fused, fp32 -> bf16), float4-vectorized
  for (int t = tid; t < KK * (IN_C / 4); t += 256) {
    int r = t / (IN_C / 4), c4 = (t % (IN_C / 4)) * 4;
    float4 v;
    if (c4 < CC)           v = *(const float4*)&hVs[i * CC + c4];
    else if (c4 < 2 * CC)  v = *(const float4*)&hE[(i * KK + r) * CC + (c4 - CC)];
    else if (IN_C == 4 * CC && c4 < 3 * CC) v = make_float4(0.f, 0.f, 0.f, 0.f);
    else                   v = *(const float4*)&hVn[nb[r] * CC + (c4 - (IN_C - CC))];
    store_bf16x4(&xs[r * IN_C + c4], v);
  }
  __syncthreads();

  // ---- GEMM1: [32,IN_C] x [IN_C,128] + b1, gelu ----
  {
    float bias = b1[n0 + n];
    for (int mt = 0; mt < 2; ++mt) {
      v8f acc;
#pragma unroll
      for (int r = 0; r < 8; ++r) acc[r] = bias;
#pragma unroll
      for (int ks = 0; ks < IN_C / 32; ++ks) {
        v16bf a = load_a_frag(xs, IN_C, 16 * mt, 32 * ks, lane);
        v16bf bb = load_b_packed(W1, 8, ks, wv, lane);
        acc = __builtin_amdgcn_wmma_f32_16x16x32_bf16(false, a, false, bb,
                                                      (short)0, acc, false, false);
      }
#pragma unroll
      for (int r = 0; r < 8; ++r)
        m1s[(16 * mt + r + 8 * h) * CC + n0 + n] = f2bf(gelu_t(acc[r]));
    }
  }
  __syncthreads();

  // ---- GEMM2: [32,128] x [128,128] + b2, gelu ----
  {
    float bias = b2[n0 + n];
    for (int mt = 0; mt < 2; ++mt) {
      v8f acc;
#pragma unroll
      for (int r = 0; r < 8; ++r) acc[r] = bias;
#pragma unroll
      for (int ks = 0; ks < 4; ++ks) {
        v16bf a = load_a_frag(m1s, CC, 16 * mt, 32 * ks, lane);
        v16bf bb = load_b_packed(W2, 8, ks, wv, lane);
        acc = __builtin_amdgcn_wmma_f32_16x16x32_bf16(false, a, false, bb,
                                                      (short)0, acc, false, false);
      }
#pragma unroll
      for (int r = 0; r < 8; ++r)
        m2s[(16 * mt + r + 8 * h) * CC + n0 + n] = f2bf(gelu_t(acc[r]));
    }
  }
  __syncthreads();

  // ---- GEMM3: [32,128] x [128,128] + b3, then masked mean or edge-LN ----
  {
    float bias = b3[n0 + n];
    float ssum = 0.0f;
    for (int mt = 0; mt < 2; ++mt) {
      v8f acc;
#pragma unroll
      for (int r = 0; r < 8; ++r) acc[r] = bias;
#pragma unroll
      for (int ks = 0; ks < 4; ++ks) {
        v16bf a = load_a_frag(m2s, CC, 16 * mt, 32 * ks, lane);
        v16bf bb = load_b_packed(W3, 8, ks, wv, lane);
        acc = __builtin_amdgcn_wmma_f32_16x16x32_bf16(false, a, false, bb,
                                                      (short)0, acc, false, false);
      }
      if (reduce) {
#pragma unroll
        for (int r = 0; r < 8; ++r) ssum += acc[r] * mrow[16 * mt + r + 8 * h];
      } else {
#pragma unroll
        for (int r = 0; r < 8; ++r)
          msf[(16 * mt + r + 8 * h) * CC + n0 + n] = acc[r];
      }
    }
    if (reduce) {
      // lanes n and n+16 hold complementary row halves of column n
      ssum += __shfl_xor(ssum, 16, 32);
      if (lane < 16) out[i * CC + n0 + lane] = ssum * (1.0f / (float)KK);
      return;  // uniform branch across whole launch
    }
  }
  __syncthreads();

  // Edge path: h_E = LN(h_E + m); each wave handles 4 edge rows (in-place safe).
  for (int q = 0; q < 4; ++q) {
    int j = wv * 4 + q;
    int cb = lane * 4;
    float4 he = *(const float4*)&hE[(i * KK + j) * CC + cb];
    float4 mm = *(const float4*)&msf[j * CC + cb];
    float xv[4] = {he.x + mm.x, he.y + mm.y, he.z + mm.z, he.w + mm.w};
    float s = xv[0] + xv[1] + xv[2] + xv[3];
    float s2 = xv[0]*xv[0] + xv[1]*xv[1] + xv[2]*xv[2] + xv[3]*xv[3];
#pragma unroll
    for (int off = 16; off; off >>= 1) {
      s  += __shfl_xor(s,  off, 32);
      s2 += __shfl_xor(s2, off, 32);
    }
    float mean = s * (1.0f / CC);
    float inv  = rsqrtf(s2 * (1.0f / CC) - mean * mean + 1e-5f);
    float4 sc = *(const float4*)&ln_s[cb];
    float4 bi = *(const float4*)&ln_b[cb];
    float4 o;
    o.x = sc.x * (xv[0] - mean) * inv + bi.x;
    o.y = sc.y * (xv[1] - mean) * inv + bi.y;
    o.z = sc.z * (xv[2] - mean) * inv + bi.z;
    o.w = sc.w * (xv[3] - mean) * inv + bi.w;
    *(float4*)&out[(i * KK + j) * CC + cb] = o;
  }
}

// Fused feed-forward: delta = gelu(hV@Wi+bi)@Wo+bo for a 16-row tile.
__global__ __launch_bounds__(256) void ff_kernel(
    const float* __restrict__ hV,
    const unsigned short* __restrict__ Wi, const float* __restrict__ bi,
    const unsigned short* __restrict__ Wo, const float* __restrict__ bo,
    float* __restrict__ out) {
  __shared__ __align__(16) unsigned short xs[16 * CC];
  __shared__ __align__(16) unsigned short hs[16 * FFH];
  const int row0 = blockIdx.x * 16;
  const int tid = threadIdx.x, lane = tid & 31, wv = tid >> 5;
  const int n = lane & 15, h = lane >> 4;

  for (int t = tid; t < 16 * CC / 4; t += 256)
    store_bf16x4(&xs[t * 4], *(const float4*)&hV[row0 * CC + t * 4]);
  __syncthreads();

  // GEMM1: [16,128] x [128,512], each wave does 4 contiguous N-tiles
  for (int tt = 0; tt < 4; ++tt) {
    int nn0 = (wv * 4 + tt) * 16;
    v8f acc;
    float bias = bi[nn0 + n];
#pragma unroll
    for (int r = 0; r < 8; ++r) acc[r] = bias;
#pragma unroll
    for (int ks = 0; ks < 4; ++ks) {
      v16bf a = load_a_frag(xs, CC, 0, 32 * ks, lane);
      v16bf bb = load_b_packed(Wi, 32, ks, wv * 4 + tt, lane);
      acc = __builtin_amdgcn_wmma_f32_16x16x32_bf16(false, a, false, bb,
                                                    (short)0, acc, false, false);
    }
#pragma unroll
    for (int r = 0; r < 8; ++r)
      hs[(r + 8 * h) * FFH + nn0 + n] = f2bf(gelu_t(acc[r]));
  }
  __syncthreads();

  // GEMM2: [16,512] x [512,128]
  {
    int nn0 = wv * 16;
    v8f acc;
    float bias = bo[nn0 + n];
#pragma unroll
    for (int r = 0; r < 8; ++r) acc[r] = bias;
#pragma unroll
    for (int ks = 0; ks < 16; ++ks) {
      v16bf a = load_a_frag(hs, FFH, 0, 32 * ks, lane);
      v16bf bb = load_b_packed(Wo, 8, ks, wv, lane);
      acc = __builtin_amdgcn_wmma_f32_16x16x32_bf16(false, a, false, bb,
                                                    (short)0, acc, false, false);
    }
#pragma unroll
    for (int r = 0; r < 8; ++r)
      out[(row0 + r + 8 * h) * CC + nn0 + n] = acc[r];
  }
}

// Residual + LayerNorm (optional mask scale). Wave per row, in-place safe.
__global__ __launch_bounds__(256) void ln_kernel(
    const float* __restrict__ base, const float* __restrict__ dlt,
    const float* __restrict__ s, const float* __restrict__ b,
    const float* __restrict__ mask, int apply_mask, float* __restrict__ out) {
  int row = blockIdx.x * 8 + (threadIdx.x >> 5);
  int lane = threadIdx.x & 31;
  int cb = lane * 4;
  float4 xb = *(const float4*)&base[row * CC + cb];
  float4 xd = *(const float4*)&dlt[row * CC + cb];
  float xv[4] = {xb.x + xd.x, xb.y + xd.y, xb.z + xd.z, xb.w + xd.w};
  float sm = xv[0] + xv[1] + xv[2] + xv[3];
  float s2 = xv[0]*xv[0] + xv[1]*xv[1] + xv[2]*xv[2] + xv[3]*xv[3];
#pragma unroll
  for (int off = 16; off; off >>= 1) {
    sm += __shfl_xor(sm, off, 32);
    s2 += __shfl_xor(s2, off, 32);
  }
  float mean = sm * (1.0f / CC);
  float inv  = rsqrtf(s2 * (1.0f / CC) - mean * mean + 1e-5f);
  float msc = apply_mask ? mask[row] : 1.0f;
  float4 sc = *(const float4*)&s[cb];
  float4 bi = *(const float4*)&b[cb];
  float4 o;
  o.x = (sc.x * (xv[0] - mean) * inv + bi.x) * msc;
  o.y = (sc.y * (xv[1] - mean) * inv + bi.y) * msc;
  o.z = (sc.z * (xv[2] - mean) * inv + bi.z) * msc;
  o.w = (sc.w * (xv[3] - mean) * inv + bi.w) * msc;
  *(float4*)&out[row * CC + cb] = o;
}

// logits = hV @ Wout + bout  ([4096,128]x[128,21] — trivially small, fp32)
__global__ __launch_bounds__(256) void out_kernel(
    const float* __restrict__ hV, const float* __restrict__ W,
    const float* __restrict__ b, float* __restrict__ out) {
  int idx = blockIdx.x * 256 + threadIdx.x;
  int row = idx >> 5, col = idx & 31;
  if (row < NN && col < AOUT) {
    float s = b[col];
#pragma unroll 8
    for (int k = 0; k < CC; ++k) s += hV[row * CC + k] * W[k * AOUT + col];
    out[row * AOUT + col] = s;
  }
}

extern "C" void kernel_launch(void* const* d_in, const int* in_sizes, int n_in,
                              void* d_out, int out_size, void* d_ws, size_t ws_size,
                              hipStream_t stream) {
  (void)in_sizes; (void)n_in; (void)out_size; (void)ws_size;

  const float* in_node  = (const float*)d_in[0];
  const float* in_edge  = (const float*)d_in[1];
  const int*   nbr      = (const int*)d_in[2];
  const float* mask     = (const float*)d_in[3];
  const float* enc_W1 = (const float*)d_in[4];   const float* enc_b1 = (const float*)d_in[5];
  const float* enc_W2 = (const float*)d_in[6];   const float* enc_b2 = (const float*)d_in[7];
  const float* enc_W3 = (const float*)d_in[8];   const float* enc_b3 = (const float*)d_in[9];
  const float* enc_n1s = (const float*)d_in[10]; const float* enc_n1b = (const float*)d_in[11];
  const float* enc_Wi = (const float*)d_in[12];  const float* enc_bi = (const float*)d_in[13];
  const float* enc_Wo = (const float*)d_in[14];  const float* enc_bo = (const float*)d_in[15];
  const float* enc_n2s = (const float*)d_in[16]; const float* enc_n2b = (const float*)d_in[17];
  const float* enc_We1 = (const float*)d_in[18]; const float* enc_be1 = (const float*)d_in[19];
  const float* enc_We2 = (const float*)d_in[20]; const float* enc_be2 = (const float*)d_in[21];
  const float* enc_We3 = (const float*)d_in[22]; const float* enc_be3 = (const float*)d_in[23];
  const float* enc_n3s = (const float*)d_in[24]; const float* enc_n3b = (const float*)d_in[25];
  const float* dec_W1 = (const float*)d_in[26];  const float* dec_b1 = (const float*)d_in[27];
  const float* dec_W2 = (const float*)d_in[28];  const float* dec_b2 = (const float*)d_in[29];
  const float* dec_W3 = (const float*)d_in[30];  const float* dec_b3 = (const float*)d_in[31];
  const float* dec_n1s = (const float*)d_in[32]; const float* dec_n1b = (const float*)d_in[33];
  const float* dec_Wi = (const float*)d_in[34];  const float* dec_bi = (const float*)d_in[35];
  const float* dec_Wo = (const float*)d_in[36];  const float* dec_bo = (const float*)d_in[37];
  const float* dec_n2s = (const float*)d_in[38]; const float* dec_n2b = (const float*)d_in[39];
  const float* w_out_W = (const float*)d_in[40]; const float* w_out_b = (const float*)d_in[41];

  // ---- workspace carve-up ----
  char* p = (char*)d_ws;
  auto alloc = [&](size_t bytes) {
    void* r = (void*)p;
    p += (bytes + 255) & ~(size_t)255;
    return r;
  };
  float* hV    = (float*)alloc(sizeof(float) * NN * CC);
  float* hVenc = (float*)alloc(sizeof(float) * NN * CC);
  float* dlt   = (float*)alloc(sizeof(float) * NN * CC);
  float* hE    = (float*)alloc(sizeof(float) * NN * KK * CC);
  auto allocw = [&](size_t nelem) { return (unsigned short*)alloc(2 * nelem); };
  unsigned short* wE1  = allocw((size_t)LEn * 3 * CC * CC);
  unsigned short* wE2  = allocw((size_t)LEn * CC * CC);
  unsigned short* wE3  = allocw((size_t)LEn * CC * CC);
  unsigned short* wEi  = allocw((size_t)LEn * CC * FFH);
  unsigned short* wEo  = allocw((size_t)LEn * FFH * CC);
  unsigned short* wEe1 = allocw((size_t)LEn * 3 * CC * CC);
  unsigned short* wEe2 = allocw((size_t)LEn * CC * CC);
  unsigned short* wEe3 = allocw((size_t)LEn * CC * CC);
  unsigned short* wD1  = allocw((size_t)LDn * 4 * CC * CC);
  unsigned short* wD2  = allocw((size_t)LDn * CC * CC);
  unsigned short* wD3  = allocw((size_t)LDn * CC * CC);
  unsigned short* wDi  = allocw((size_t)LDn * CC * FFH);
  unsigned short* wDo  = allocw((size_t)LDn * FFH * CC);

  auto pack = [&](const float* s, unsigned short* d, int Kd, int NC, int layers) {
    int total = layers * Kd * NC;
    pack_b_kernel<<<(total + 255) / 256, 256, 0, stream>>>(s, d, Kd, NC, layers);
  };
  pack(enc_W1,  wE1,  3 * CC, CC,  LEn);
  pack(enc_W2,  wE2,  CC,     CC,  LEn);
  pack(enc_W3,  wE3,  CC,     CC,  LEn);
  pack(enc_Wi,  wEi,  CC,     FFH, LEn);
  pack(enc_Wo,  wEo,  FFH,    CC,  LEn);
  pack(enc_We1, wEe1, 3 * CC, CC,  LEn);
  pack(enc_We2, wEe2, CC,     CC,  LEn);
  pack(enc_We3, wEe3, CC,     CC,  LEn);
  pack(dec_W1,  wD1,  4 * CC, CC,  LDn);
  pack(dec_W2,  wD2,  CC,     CC,  LDn);
  pack(dec_W3,  wD3,  CC,     CC,  LDn);
  pack(dec_Wi,  wDi,  CC,     FFH, LDn);
  pack(dec_Wo,  wDo,  FFH,    CC,  LDn);

  hipMemcpyAsync(hV, in_node, sizeof(float) * NN * CC, hipMemcpyDeviceToDevice, stream);
  hipMemcpyAsync(hE, in_edge, sizeof(float) * NN * KK * CC, hipMemcpyDeviceToDevice, stream);

  // ---------------- Encoder ----------------
  for (int l = 0; l < LEn; ++l) {
    msg_kernel<3 * CC><<<NN, 256, 0, stream>>>(
        hV, hV, hE, nbr, mask,
        wE1 + (size_t)l * 3 * CC * CC, enc_b1 + l * CC,
        wE2 + (size_t)l * CC * CC,     enc_b2 + l * CC,
        wE3 + (size_t)l * CC * CC,     enc_b3 + l * CC,
        nullptr, nullptr, dlt, 1);
    ln_kernel<<<NN / 8, 256, 0, stream>>>(hV, dlt, enc_n1s + l * CC, enc_n1b + l * CC,
                                          mask, 0, hV);
    ff_kernel<<<NN / 16, 256, 0, stream>>>(hV, wEi + (size_t)l * CC * FFH, enc_bi + l * FFH,
                                           wEo + (size_t)l * FFH * CC, enc_bo + l * CC, dlt);
    ln_kernel<<<NN / 8, 256, 0, stream>>>(hV, dlt, enc_n2s + l * CC, enc_n2b + l * CC,
                                          mask, 1, hV);
    msg_kernel<3 * CC><<<NN, 256, 0, stream>>>(
        hV, hV, hE, nbr, mask,
        wEe1 + (size_t)l * 3 * CC * CC, enc_be1 + l * CC,
        wEe2 + (size_t)l * CC * CC,     enc_be2 + l * CC,
        wEe3 + (size_t)l * CC * CC,     enc_be3 + l * CC,
        enc_n3s + l * CC, enc_n3b + l * CC, hE, 0);
  }

  hipMemcpyAsync(hVenc, hV, sizeof(float) * NN * CC, hipMemcpyDeviceToDevice, stream);

  // ---------------- Decoder ----------------
  for (int l = 0; l < LDn; ++l) {
    msg_kernel<4 * CC><<<NN, 256, 0, stream>>>(
        hV, hVenc, hE, nbr, mask,
        wD1 + (size_t)l * 4 * CC * CC, dec_b1 + l * CC,
        wD2 + (size_t)l * CC * CC,     dec_b2 + l * CC,
        wD3 + (size_t)l * CC * CC,     dec_b3 + l * CC,
        nullptr, nullptr, dlt, 1);
    ln_kernel<<<NN / 8, 256, 0, stream>>>(hV, dlt, dec_n1s + l * CC, dec_n1b + l * CC,
                                          mask, 0, hV);
    ff_kernel<<<NN / 16, 256, 0, stream>>>(hV, wDi + (size_t)l * CC * FFH, dec_bi + l * FFH,
                                           wDo + (size_t)l * FFH * CC, dec_bo + l * CC, dlt);
    ln_kernel<<<NN / 8, 256, 0, stream>>>(hV, dlt, dec_n2s + l * CC, dec_n2b + l * CC,
                                          mask, 1, hV);
  }

  out_kernel<<<(NN * 32) / 256, 256, 0, stream>>>(hV, w_out_W, w_out_b, (float*)d_out);
}
